// SplitTransformer_36636071035466
// MI455X (gfx1250) — matrix operand order repaired and checked by reading, not compiled, not run on hardware
//
#include <hip/hip_runtime.h>
#include <math.h>

// ---------------- geometry constants (from reference) ----------------
#define SEQ_LEN 4096
#define CHUNK   512
#define RANGE   128
#define DMODEL  256
#define NHEAD   8
#define DHEAD   32
#define FFDIM   1024
#define NC      8
#define BATCH   4
#define MAXW    768
#define LN_EPS  1e-3f

typedef _Float16 half_t;
typedef __attribute__((ext_vector_type(8)))  _Float16 v8h;
typedef __attribute__((ext_vector_type(16))) _Float16 v16h;
typedef __attribute__((ext_vector_type(8)))  float    v8f;

__device__ __forceinline__ v16h cat8(v8h lo, v8h hi) {
    v16h r;
#pragma unroll
    for (int t = 0; t < 8; ++t) { r[t] = lo[t]; r[t + 8] = hi[t]; }
    return r;
}
__device__ __forceinline__ v8f zero8f() {
    v8f r;
#pragma unroll
    for (int t = 0; t < 8; ++t) r[t] = 0.f;
    return r;
}

// ---------------- embed + positional ----------------
__global__ __launch_bounds__(256)
void embed_kernel(const int* __restrict__ toks, const float* __restrict__ emb,
                  const float* __restrict__ pos, float* __restrict__ hF,
                  half_t* __restrict__ hH, int W, int ms, int ci) {
    int t = blockIdx.x * 256 + threadIdx.x;
    int total = BATCH * W * DMODEL;
    if (t >= total) return;
    int d = t & 255;
    int w = (t >> 8) % W;
    int bb = t / (W * DMODEL);
    int tok = toks[bb * SEQ_LEN + ms + w];
    float val = emb[tok * DMODEL + d] + pos[((size_t)ci * MAXW + w) * DMODEL + d];
    hF[t] = val;
    hH[t] = (half_t)val;
}

// ---------------- weight convert + transpose: [b][r][c] f32 -> [b][c][r] f16 ----------------
__global__ __launch_bounds__(256)
void wtrans_kernel(const float* __restrict__ in, half_t* __restrict__ out,
                   int Rr, int Cc) {
    long t = (long)blockIdx.x * 256 + threadIdx.x;
    long total = (long)NC * Rr * Cc;
    if (t >= total) return;
    int c = (int)(t % Cc);
    long rb = t / Cc;
    int r = (int)(rb % Rr);
    int bb = (int)(rb / Rr);
    out[((size_t)bb * Cc + c) * Rr + r] = (half_t)in[t];
}

// ---------------- V transpose: [b*W+w][h*32+dh] -> [b][h][dh][w] ----------------
__global__ __launch_bounds__(256)
void vtrans_kernel(const half_t* __restrict__ v, half_t* __restrict__ vT, int W) {
    int t = blockIdx.x * 256 + threadIdx.x;
    int total = BATCH * W * DMODEL;
    if (t >= total) return;
    int d = t & 255;
    int w = (t >> 8) % W;
    int bb = t / (W * DMODEL);
    int hh = d >> 5, dh = d & 31;
    vT[(((size_t)bb * NHEAD + hh) * DHEAD + dh) * W + w] = v[t];
}

// ---------------- generic WMMA GEMM: out = act(A[M,K] @ Bt[N,K]^T + bias) ----------------
// act: 0 = none, 1 = exact gelu
__global__ __launch_bounds__(256)
void gemm_wmma(const half_t* __restrict__ A, const half_t* __restrict__ Bt,
               const float* __restrict__ bias, float* __restrict__ outF,
               half_t* __restrict__ outH, int M, int N, int K, int act) {
    int lane = threadIdx.x & 31;
    int wave = threadIdx.x >> 5;
    int l15 = lane & 15, lg = lane >> 4;
    int wm = wave & 3, wn = wave >> 2;              // 4 x 2 wave grid
    int m0 = blockIdx.x * 128 + wm * 32;            // 32 rows per wave
    int n0 = blockIdx.y * 128 + wn * 64;            // 64 cols per wave

    v8f acc[2][4];
#pragma unroll
    for (int i = 0; i < 2; ++i)
#pragma unroll
        for (int j = 0; j < 4; ++j) acc[i][j] = zero8f();

    for (int k0 = 0; k0 < K; k0 += 32) {
        v16h a[2];
#pragma unroll
        for (int i = 0; i < 2; ++i) {
            const half_t* ap = A + (size_t)(m0 + 16 * i + l15) * K + k0 + 16 * lg;
            a[i] = cat8(*(const v8h*)ap, *(const v8h*)(ap + 8));
        }
#pragma unroll
        for (int j = 0; j < 4; ++j) {
            const half_t* bp = Bt + (size_t)(n0 + 16 * j + l15) * K + k0 + 16 * lg;
            v16h b = cat8(*(const v8h*)bp, *(const v8h*)(bp + 8));
#pragma unroll
            for (int i = 0; i < 2; ++i)
                acc[i][j] = __builtin_amdgcn_wmma_f32_16x16x32_f16(
                    false, a[i], false, b, (short)0, acc[i][j], false, false);
        }
    }

#pragma unroll
    for (int j = 0; j < 4; ++j) {
        int col = n0 + 16 * j + l15;
        float bv = bias[col];
#pragma unroll
        for (int i = 0; i < 2; ++i) {
#pragma unroll
            for (int r = 0; r < 8; ++r) {
                float x = acc[i][j][r] + bv;
                if (act == 1) x = 0.5f * x * (1.0f + erff(x * 0.70710678118654752f));
                int row = m0 + 16 * i + r + 8 * lg;
                size_t o = (size_t)row * N + col;
                if (outF) outF[o] = x;
                if (outH) outH[o] = (half_t)x;
            }
        }
    }
}

// ---------------- banded attention, flash-style, one wave / 16-query tile ----------------
__global__ __launch_bounds__(32)
void attn_kernel(const half_t* __restrict__ Q, const half_t* __restrict__ Kf,
                 const half_t* __restrict__ Vt, half_t* __restrict__ O, int W) {
    __shared__ half_t pbuf[16 * 32];
    int lane = threadIdx.x;
    int l15 = lane & 15, lg = lane >> 4;
    int q0 = blockIdx.x * 16;
    int hh = blockIdx.y;
    int bb = blockIdx.z;

    // Q fragment (A operand, K-dim = DHEAD = 32)
    const half_t* qp = Q + ((size_t)(bb * W) + q0 + l15) * DMODEL + hh * DHEAD + 16 * lg;
    v16h qa = cat8(*(const v8h*)qp, *(const v8h*)(qp + 8));

    float mrow[8], lrow[8];
    v8f acc0 = zero8f(), acc1 = zero8f();
#pragma unroll
    for (int r = 0; r < 8; ++r) { mrow[r] = -3.0e38f; lrow[r] = 0.f; }

    int kstart = (q0 > RANGE ? q0 - RANGE : 0) & ~31;
    int kend = q0 + 16 + RANGE; if (kend > W) kend = W;

    for (int kk = kstart; kk < kend; kk += 32) {
        v8f s[2];
#pragma unroll
        for (int t = 0; t < 2; ++t) {
            int key = kk + 16 * t + l15;
            int keyc = key < (W - 1) ? key : (W - 1);
            const half_t* kp = Kf + ((size_t)(bb * W) + keyc) * DMODEL + hh * DHEAD + 16 * lg;
            v16h kb = cat8(*(const v8h*)kp, *(const v8h*)(kp + 8));
            v8f z = zero8f();
            s[t] = __builtin_amdgcn_wmma_f32_16x16x32_f16(false, qa, false, kb,
                                                          (short)0, z, false, false);
        }
        // scale + band mask + online softmax update
#pragma unroll
        for (int r = 0; r < 8; ++r) {
            int q = q0 + r + 8 * lg;
            float best = -3.0e38f;
#pragma unroll
            for (int t = 0; t < 2; ++t) {
                int key = kk + 16 * t + l15;
                float sc = s[t][r] * 0.17677669529663687f;
                int diff = q - key; if (diff < 0) diff = -diff;
                bool ok = (key < kend) && (diff <= RANGE);
                sc = ok ? sc : -1.0e9f;
                s[t][r] = sc;
                best = fmaxf(best, sc);
            }
            best = fmaxf(best, __shfl_xor(best, 1));
            best = fmaxf(best, __shfl_xor(best, 2));
            best = fmaxf(best, __shfl_xor(best, 4));
            best = fmaxf(best, __shfl_xor(best, 8));
            float mnew = fmaxf(mrow[r], best);
            float alpha = __expf(mrow[r] - mnew);
            mrow[r] = mnew;
            float psum = 0.f;
#pragma unroll
            for (int t = 0; t < 2; ++t) {
                float p = __expf(s[t][r] - mnew);
                psum += p;
                pbuf[(r + 8 * lg) * 32 + 16 * t + l15] = (half_t)p;
            }
            psum += __shfl_xor(psum, 1);
            psum += __shfl_xor(psum, 2);
            psum += __shfl_xor(psum, 4);
            psum += __shfl_xor(psum, 8);
            lrow[r] = lrow[r] * alpha + psum;
            acc0[r] *= alpha;
            acc1[r] *= alpha;
        }
        __syncthreads();
        // probs fragment (A operand, K-dim = 32 keys)
        const half_t* pp = &pbuf[l15 * 32 + 16 * lg];
        v16h pa = cat8(*(const v8h*)pp, *(const v8h*)(pp + 8));
        // V fragments (B operand): dh columns, keys contiguous via Vt layout
        int kb0 = kk + 16 * lg;
        if (kb0 > W - 16) kb0 = W - 16;
#pragma unroll
        for (int t = 0; t < 2; ++t) {
            const half_t* vp =
                Vt + (((size_t)bb * NHEAD + hh) * DHEAD + 16 * t + l15) * W + kb0;
            v16h vb = cat8(*(const v8h*)vp, *(const v8h*)(vp + 8));
            if (t == 0)
                acc0 = __builtin_amdgcn_wmma_f32_16x16x32_f16(false, pa, false, vb,
                                                              (short)0, acc0, false, false);
            else
                acc1 = __builtin_amdgcn_wmma_f32_16x16x32_f16(false, pa, false, vb,
                                                              (short)0, acc1, false, false);
        }
        __syncthreads();
    }

#pragma unroll
    for (int r = 0; r < 8; ++r) {
        int row = q0 + r + 8 * lg;
        float inv = 1.0f / lrow[r];
        size_t base = ((size_t)(bb * W) + row) * DMODEL + hh * DHEAD;
        O[base + l15] = (half_t)(acc0[r] * inv);
        O[base + 16 + l15] = (half_t)(acc1[r] * inv);
    }
}

// ---------------- residual + LayerNorm (1 wave per row of 256) ----------------
__global__ __launch_bounds__(256)
void ln_kernel(const float* __restrict__ X, const float* __restrict__ R,
               const float* __restrict__ g, const float* __restrict__ b,
               float* __restrict__ outF, half_t* __restrict__ outH, int Mrows) {
    int lane = threadIdx.x & 31;
    int wv = threadIdx.x >> 5;
    int row = blockIdx.x * 8 + wv;
    if (row >= Mrows) return;
    const float* x = X + (size_t)row * DMODEL;
    const float* rr = R + (size_t)row * DMODEL;
    float v[8];
    float s = 0.f;
#pragma unroll
    for (int j = 0; j < 8; ++j) { v[j] = x[lane + 32 * j] + rr[lane + 32 * j]; s += v[j]; }
    s += __shfl_xor(s, 1); s += __shfl_xor(s, 2); s += __shfl_xor(s, 4);
    s += __shfl_xor(s, 8); s += __shfl_xor(s, 16);
    float mean = s * (1.0f / DMODEL);
    float q = 0.f;
#pragma unroll
    for (int j = 0; j < 8; ++j) { float d = v[j] - mean; q += d * d; }
    q += __shfl_xor(q, 1); q += __shfl_xor(q, 2); q += __shfl_xor(q, 4);
    q += __shfl_xor(q, 8); q += __shfl_xor(q, 16);
    float rs = rsqrtf(q * (1.0f / DMODEL) + LN_EPS);
#pragma unroll
    for (int j = 0; j < 8; ++j) {
        int col = lane + 32 * j;
        float y = (v[j] - mean) * rs * g[col] + b[col];
        size_t o = (size_t)row * DMODEL + col;
        if (outF) outF[o] = y;
        if (outH) outH[o] = (half_t)y;
    }
}

// ---------------- final LN + slice into d_out ----------------
__global__ __launch_bounds__(256)
void ln_out_kernel(const float* __restrict__ X, const float* __restrict__ R,
                   const float* __restrict__ g, const float* __restrict__ b,
                   float* __restrict__ out, int W, int ms, int so) {
    int lane = threadIdx.x & 31;
    int wv = threadIdx.x >> 5;
    int rid = blockIdx.x * 8 + wv;          // rid in [0, BATCH*CHUNK)
    if (rid >= BATCH * CHUNK) return;
    int bb = rid >> 9;
    int w = so + (rid & 511);
    int row = bb * W + w;
    const float* x = X + (size_t)row * DMODEL;
    const float* rr = R + (size_t)row * DMODEL;
    float v[8];
    float s = 0.f;
#pragma unroll
    for (int j = 0; j < 8; ++j) { v[j] = x[lane + 32 * j] + rr[lane + 32 * j]; s += v[j]; }
    s += __shfl_xor(s, 1); s += __shfl_xor(s, 2); s += __shfl_xor(s, 4);
    s += __shfl_xor(s, 8); s += __shfl_xor(s, 16);
    float mean = s * (1.0f / DMODEL);
    float q = 0.f;
#pragma unroll
    for (int j = 0; j < 8; ++j) { float d = v[j] - mean; q += d * d; }
    q += __shfl_xor(q, 1); q += __shfl_xor(q, 2); q += __shfl_xor(q, 4);
    q += __shfl_xor(q, 8); q += __shfl_xor(q, 16);
    float rs = rsqrtf(q * (1.0f / DMODEL) + LN_EPS);
#pragma unroll
    for (int j = 0; j < 8; ++j) {
        int col = lane + 32 * j;
        float y = (v[j] - mean) * rs * g[col] + b[col];
        out[((size_t)bb * SEQ_LEN + ms + w) * DMODEL + col] = y;
    }
}

// ---------------- host launcher ----------------
extern "C" void kernel_launch(void* const* d_in, const int* in_sizes, int n_in,
                              void* d_out, int out_size, void* d_ws, size_t ws_size,
                              hipStream_t stream) {
    const int*   toks  = (const int*)d_in[0];
    const float* emb   = (const float*)d_in[1];
    const float* pos   = (const float*)d_in[2];
    const float* Wq    = (const float*)d_in[3];
    const float* bq    = (const float*)d_in[4];
    const float* Wk    = (const float*)d_in[5];
    const float* bk    = (const float*)d_in[6];
    const float* Wv    = (const float*)d_in[7];
    const float* bv    = (const float*)d_in[8];
    const float* Wo    = (const float*)d_in[9];
    const float* bo    = (const float*)d_in[10];
    const float* ln1g  = (const float*)d_in[11];
    const float* ln1b  = (const float*)d_in[12];
    const float* W1    = (const float*)d_in[13];
    const float* b1    = (const float*)d_in[14];
    const float* W2    = (const float*)d_in[15];
    const float* b2    = (const float*)d_in[16];
    const float* ln2g  = (const float*)d_in[17];
    const float* ln2b  = (const float*)d_in[18];
    float* out = (float*)d_out;

    char* ws = (char*)d_ws;
    size_t off = 0;
    auto alloc = [&](size_t bytes) -> void* {
        void* p = ws + off;
        off += (bytes + 255) & ~(size_t)255;
        return p;
    };

    const int MMAX = BATCH * MAXW;  // 3072
    half_t* WqT = (half_t*)alloc((size_t)NC * DMODEL * DMODEL * 2);
    half_t* WkT = (half_t*)alloc((size_t)NC * DMODEL * DMODEL * 2);
    half_t* WvT = (half_t*)alloc((size_t)NC * DMODEL * DMODEL * 2);
    half_t* WoT = (half_t*)alloc((size_t)NC * DMODEL * DMODEL * 2);
    half_t* W1T = (half_t*)alloc((size_t)NC * FFDIM * DMODEL * 2);
    half_t* W2T = (half_t*)alloc((size_t)NC * DMODEL * FFDIM * 2);
    float*  hF  = (float*)alloc((size_t)MMAX * DMODEL * 4);
    half_t* hH  = (half_t*)alloc((size_t)MMAX * DMODEL * 2);
    half_t* qH  = (half_t*)alloc((size_t)MMAX * DMODEL * 2);
    half_t* kH  = (half_t*)alloc((size_t)MMAX * DMODEL * 2);
    half_t* vH  = (half_t*)alloc((size_t)MMAX * DMODEL * 2);
    half_t* vT  = (half_t*)alloc((size_t)MMAX * DMODEL * 2);
    half_t* aH  = (half_t*)alloc((size_t)MMAX * DMODEL * 2);
    float*  aoF = (float*)alloc((size_t)MMAX * DMODEL * 4);
    float*  h1F = (float*)alloc((size_t)MMAX * DMODEL * 4);
    half_t* h1H = (half_t*)alloc((size_t)MMAX * DMODEL * 2);
    half_t* gH  = (half_t*)alloc((size_t)MMAX * FFDIM * 2);
    float*  f2F = (float*)alloc((size_t)MMAX * DMODEL * 4);

    // one-time weight convert+transpose (per call; cheap, graph-safe)
    {
        long tDD = (long)NC * DMODEL * DMODEL;
        int gDD = (int)((tDD + 255) / 256);
        wtrans_kernel<<<gDD, 256, 0, stream>>>(Wq, WqT, DMODEL, DMODEL);
        wtrans_kernel<<<gDD, 256, 0, stream>>>(Wk, WkT, DMODEL, DMODEL);
        wtrans_kernel<<<gDD, 256, 0, stream>>>(Wv, WvT, DMODEL, DMODEL);
        wtrans_kernel<<<gDD, 256, 0, stream>>>(Wo, WoT, DMODEL, DMODEL);
        long tDF = (long)NC * DMODEL * FFDIM;
        int gDF = (int)((tDF + 255) / 256);
        wtrans_kernel<<<gDF, 256, 0, stream>>>(W1, W1T, DMODEL, FFDIM);  // -> [FF, D]
        wtrans_kernel<<<gDF, 256, 0, stream>>>(W2, W2T, FFDIM, DMODEL);  // -> [D, FF]
    }

    for (int i = 0; i < NC; ++i) {
        int cs = i * CHUNK;
        int ce = cs + CHUNK;
        int ms = cs - RANGE; if (ms < 0) ms = 0;
        int me = ce + RANGE; if (me > SEQ_LEN) me = SEQ_LEN;
        int W = me - ms;
        int so = cs - ms;
        int M = BATCH * W;

        int elemBlocks = (M * DMODEL) / 256;
        embed_kernel<<<elemBlocks, 256, 0, stream>>>(toks, emb, pos, hF, hH, W, ms, i);

        dim3 gD(M / 128, DMODEL / 128);
        gemm_wmma<<<gD, 256, 0, stream>>>(hH, WqT + (size_t)i * DMODEL * DMODEL,
                                          bq + i * DMODEL, nullptr, qH, M, DMODEL, DMODEL, 0);
        gemm_wmma<<<gD, 256, 0, stream>>>(hH, WkT + (size_t)i * DMODEL * DMODEL,
                                          bk + i * DMODEL, nullptr, kH, M, DMODEL, DMODEL, 0);
        gemm_wmma<<<gD, 256, 0, stream>>>(hH, WvT + (size_t)i * DMODEL * DMODEL,
                                          bv + i * DMODEL, nullptr, vH, M, DMODEL, DMODEL, 0);

        vtrans_kernel<<<elemBlocks, 256, 0, stream>>>(vH, vT, W);

        dim3 gA(W / 16, NHEAD, BATCH);
        attn_kernel<<<gA, 32, 0, stream>>>(qH, kH, vT, aH, W);

        gemm_wmma<<<gD, 256, 0, stream>>>(aH, WoT + (size_t)i * DMODEL * DMODEL,
                                          bo + i * DMODEL, aoF, nullptr, M, DMODEL, DMODEL, 0);

        ln_kernel<<<M / 8, 256, 0, stream>>>(aoF, hF, ln1g + i * DMODEL, ln1b + i * DMODEL,
                                             h1F, h1H, M);

        dim3 gF1(M / 128, FFDIM / 128);
        gemm_wmma<<<gF1, 256, 0, stream>>>(h1H, W1T + (size_t)i * FFDIM * DMODEL,
                                           b1 + i * FFDIM, nullptr, gH, M, FFDIM, DMODEL, 1);

        gemm_wmma<<<gD, 256, 0, stream>>>(gH, W2T + (size_t)i * DMODEL * FFDIM,
                                          b2 + i * DMODEL, f2F, nullptr, M, DMODEL, FFDIM, 0);

        ln_out_kernel<<<(BATCH * CHUNK) / 8, 256, 0, stream>>>(
            f2F, h1F, ln2g + i * DMODEL, ln2b + i * DMODEL, out, W, ms, so);
    }
}
